// Temporal_VQVAE_66984309948647
// MI455X (gfx1250) — matrix operand order repaired and checked by reading, not compiled, not run on hardware
//
#include <hip/hip_runtime.h>
#include <hip/hip_bf16.h>

// ---------------------------------------------------------------------------
// Temporal VQ-VAE forward for MI455X (gfx1250, wave32, WMMA).
// All dense contractions -> v_wmma_f32_16x16x32_f16 (f16 in, f32 accum).
// Weights/codebook pre-packed once per call into the exact WMMA B-fragment
// lane layout (f16), so each lane fetches its fragment with 2x b128 loads.
// A operands fetched as float4 (b128) + v_cvt_pk_f16_f32.
// VQ nearest-code search fused into the distance GEMM (no 267MB dist buffer).
// ---------------------------------------------------------------------------

typedef __attribute__((ext_vector_type(16))) _Float16 v16h;
typedef __attribute__((ext_vector_type(8)))  float    v8f;
typedef __attribute__((ext_vector_type(4)))  float    v4f;

#define B_    16
#define T_IN_ 65536
#define T1_   4081          // (65536-256)/16 + 1
#define T2_   1020          // (4081-4)/4 + 1
#define T3_   4080          // (1020-1)*4 + 4
#define TOUT_ 65520         // (4080-1)*16 + 256
#define EMB_  64
#define FF_   256
#define CB_   4096
#define N1_   (B_*T1_)      // 65296 rows (conv1 GEMM)
#define NTOK_ (B_*T2_)      // 16320 tokens
#define LOG10K_ 9.210340371976184f
#define LSTRIDE_ 49152      // packed halves per transformer layer

__device__ __forceinline__ float gelu_f(float x) {
  return 0.5f * x * (1.0f + erff(x * 0.70710678118654752f));
}

// ---------------------------------------------------------------------------
// Pack a KxN f32 matrix (optionally given as NxK, i.e. transposed source) into
// f16 WMMA B-fragment layout:
//   P[(((nt*(K/32)+kc)*32 + lane)*16 + e],  n = nt*16 + (lane&15),
//   k = kc*32 + (e<8 ? 8*(lane>>4) + e : 16 + 8*(lane>>4) + (e-8))
// so each lane's 16-half fragment is one contiguous 32-byte chunk.
// ---------------------------------------------------------------------------
__global__ void pack_b16(const float* __restrict__ W, _Float16* __restrict__ P,
                         int K, int N, int transposed)
{
  int i = blockIdx.x * 256 + threadIdx.x;
  if (i >= K * N) return;
  int k = i / N, n = i % N;
  float v = transposed ? W[(long)n * K + k] : W[(long)k * N + n];
  int nt = n >> 4, cl = n & 15, kc = k >> 5, kr = k & 31;
  int lane = ((kr >> 3) & 1) ? (cl + 16) : cl;
  int e = (kr & 7) + ((kr >> 4) << 3);
  P[(((size_t)nt * (K >> 5) + kc) * 32 + lane) * 16 + e] = (_Float16)v;
}

// ---------------------------------------------------------------------------
// Generic 16x16-tile GEMM, one wave32 per C tile, K multiple of 32.
//   A row r lives at  A + a_mul*r + a_extra*(r / rpb)        (implicit im2col)
//   C/resid row r_out = out_mul*r + out_add                  (tap interleave)
// ---------------------------------------------------------------------------
__global__ __launch_bounds__(32) void gemm16(
    const float* __restrict__ A, const _Float16* __restrict__ Bp,
    const float* __restrict__ bias, const float* __restrict__ resid,
    float* __restrict__ C,
    int K, int ldc,
    int a_mul, int a_extra, int rpb,
    int out_mul, int out_add, int act_gelu)
{
  const int lane = threadIdx.x;
  const int mt = blockIdx.x, nt = blockIdx.y;
  const int rowA = mt * 16 + (lane & 15);
  const float* Ap = A + (long)a_mul * rowA + (long)a_extra * (rowA / rpb);
  const int kA = (lane < 16) ? 0 : 8;
  const int col = nt * 16 + (lane & 15);
  const int KT = K >> 5;
  const _Float16* Bt = Bp + ((size_t)nt * KT * 32 + lane) * 16;

  v8f acc = {0.f, 0.f, 0.f, 0.f, 0.f, 0.f, 0.f, 0.f};
  for (int kc = 0; kc < KT; ++kc) {
    const int kk = kc << 5;
    if (kc + 2 < KT) __builtin_prefetch(Ap + kk + 64, 0, 0);   // global_prefetch_b8
    const v4f x0 = *(const v4f*)(Ap + kk + kA);
    const v4f x1 = *(const v4f*)(Ap + kk + kA + 4);
    const v4f x2 = *(const v4f*)(Ap + kk + kA + 16);
    const v4f x3 = *(const v4f*)(Ap + kk + kA + 20);
    v16h a;
#pragma unroll
    for (int e = 0; e < 4; ++e) {
      a[e]      = (_Float16)x0[e];
      a[e + 4]  = (_Float16)x1[e];
      a[e + 8]  = (_Float16)x2[e];
      a[e + 12] = (_Float16)x3[e];
    }
    const v16h b = *(const v16h*)(Bt + (size_t)kc * 512);      // 2x b128
    acc = __builtin_amdgcn_wmma_f32_16x16x32_f16(false, a, false, b,
                                                 (short)0, acc, false, false);
  }
  const int rbase = mt * 16 + ((lane < 16) ? 0 : 8);
  const float bi = bias ? bias[col] : 0.f;
#pragma unroll
  for (int j = 0; j < 8; ++j) {
    long orow = (long)out_mul * (rbase + j) + out_add;
    float v = acc[j] + bi;
    if (resid) v += resid[orow * ldc + col];
    if (act_gelu) v = gelu_f(v);
    C[orow * ldc + col] = v;
  }
}

// ---------------------------------------------------------------------------
// Fused VQ: one wave per 16-row tile of quant_in; streams 256 x 16-code tiles
// (packed f16 codebook) through WMMA, tracks argmin of 0.5*||c||^2 - x.c,
// then cross-lane reduces with __shfl_xor (wave32).
// ---------------------------------------------------------------------------
__global__ __launch_bounds__(32) void vq_argmin(
    const float* __restrict__ X, const _Float16* __restrict__ cbh,
    const float* __restrict__ cbn2, int* __restrict__ idx_i,
    float* __restrict__ idx_f)
{
  const int lane = threadIdx.x, mt = blockIdx.x;
  const float* Xp = X + (long)(mt * 16 + (lane & 15)) * EMB_;
  const int kA = (lane < 16) ? 0 : 8;
  v16h a0, a1;
#pragma unroll
  for (int h = 0; h < 2; ++h) {
    const float* xp = Xp + h * 32;
    const v4f x0 = *(const v4f*)(xp + kA);
    const v4f x1 = *(const v4f*)(xp + kA + 4);
    const v4f x2 = *(const v4f*)(xp + kA + 16);
    const v4f x3 = *(const v4f*)(xp + kA + 20);
#pragma unroll
    for (int e = 0; e < 4; ++e) {
      if (h == 0) {
        a0[e] = (_Float16)x0[e]; a0[e + 4] = (_Float16)x1[e];
        a0[e + 8] = (_Float16)x2[e]; a0[e + 12] = (_Float16)x3[e];
      } else {
        a1[e] = (_Float16)x0[e]; a1[e + 4] = (_Float16)x1[e];
        a1[e + 8] = (_Float16)x2[e]; a1[e + 12] = (_Float16)x3[e];
      }
    }
  }
  const _Float16* Ct = cbh + (size_t)lane * 16;
  float best[8]; int bidx[8];
#pragma unroll
  for (int j = 0; j < 8; ++j) { best[j] = 3.4e38f; bidx[j] = 0; }

  for (int ct = 0; ct < CB_ / 16; ++ct) {
    const int c = ct * 16 + (lane & 15);
    const v16h b0 = *(const v16h*)(Ct + (size_t)ct * 1024);
    const v16h b1 = *(const v16h*)(Ct + (size_t)ct * 1024 + 512);
    v8f acc = {0.f, 0.f, 0.f, 0.f, 0.f, 0.f, 0.f, 0.f};
    acc = __builtin_amdgcn_wmma_f32_16x16x32_f16(false, a0, false, b0, (short)0, acc, false, false);
    acc = __builtin_amdgcn_wmma_f32_16x16x32_f16(false, a1, false, b1, (short)0, acc, false, false);
    const float hn = cbn2[c];
#pragma unroll
    for (int j = 0; j < 8; ++j) {
      float d = hn - acc[j];
      if (d < best[j]) { best[j] = d; bidx[j] = c; }
    }
  }
#pragma unroll
  for (int m = 1; m < 16; m <<= 1) {
#pragma unroll
    for (int j = 0; j < 8; ++j) {
      float ov = __shfl_xor(best[j], m, 32);
      int   oi = __shfl_xor(bidx[j], m, 32);
      if (ov < best[j] || (ov == best[j] && oi < bidx[j])) { best[j] = ov; bidx[j] = oi; }
    }
  }
  if ((lane & 15) == 0) {
    const int rb = mt * 16 + ((lane < 16) ? 0 : 8);
#pragma unroll
    for (int j = 0; j < 8; ++j) {
      idx_i[rb + j] = bidx[j];
      idx_f[rb + j] = (float)bidx[j];
    }
  }
}

// ------------------------- elementwise / reduction helpers ------------------
__global__ __launch_bounds__(32) void layernorm_row(
    const float* __restrict__ x, float* __restrict__ y,
    const float* __restrict__ w, const float* __restrict__ b)
{
  const int n = blockIdx.x, lane = threadIdx.x;
  const float* xp = x + (long)n * EMB_;
  float v0 = xp[lane], v1 = xp[lane + 32];
  float s = v0 + v1, q = v0 * v0 + v1 * v1;
  for (int m = 1; m < 32; m <<= 1) { s += __shfl_xor(s, m, 32); q += __shfl_xor(q, m, 32); }
  float mean = s * (1.f / EMB_);
  float inv = rsqrtf(q * (1.f / EMB_) - mean * mean + 1e-5f);
  float* yp = y + (long)n * EMB_;
  yp[lane]      = (v0 - mean) * inv * w[lane]      + b[lane];
  yp[lane + 32] = (v1 - mean) * inv * w[lane + 32] + b[lane + 32];
}

__global__ __launch_bounds__(32) void l2norm_row(
    const float* __restrict__ x, float* __restrict__ y)
{
  const int n = blockIdx.x, lane = threadIdx.x;
  const float* xp = x + (long)n * EMB_;
  float v0 = xp[lane], v1 = xp[lane + 32];
  float q = v0 * v0 + v1 * v1;
  for (int m = 1; m < 32; m <<= 1) q += __shfl_xor(q, m, 32);
  float inv = rsqrtf(q + 1e-12f);
  float* yp = y + (long)n * EMB_;
  yp[lane] = v0 * inv;  yp[lane + 32] = v1 * inv;
}

__global__ void groupnorm_tok(float* __restrict__ buf, const float* __restrict__ w,
                              const float* __restrict__ b, int T)
{
  __shared__ float ssum[256], ssq[256];
  const int bg = blockIdx.x, bb = bg >> 4, g = bg & 15;
  const long base = (long)bb * T * EMB_ + g * 4;
  float s = 0.f, q = 0.f;
  for (int i = threadIdx.x; i < T * 4; i += 256) {
    float v = buf[base + (long)(i >> 2) * EMB_ + (i & 3)];
    s += v; q += v * v;
  }
  ssum[threadIdx.x] = s; ssq[threadIdx.x] = q; __syncthreads();
  for (int st = 128; st > 0; st >>= 1) {
    if (threadIdx.x < st) { ssum[threadIdx.x] += ssum[threadIdx.x + st]; ssq[threadIdx.x] += ssq[threadIdx.x + st]; }
    __syncthreads();
  }
  const float mean = ssum[0] / (T * 4.f);
  const float inv  = rsqrtf(ssq[0] / (T * 4.f) - mean * mean + 1e-5f);
  for (int i = threadIdx.x; i < T * 4; i += 256) {
    int cc = i & 3, c = g * 4 + cc;
    long o = base + (long)(i >> 2) * EMB_ + cc;
    buf[o] = (buf[o] - mean) * inv * w[c] + b[c];
  }
}

__global__ void softmax_head(float* __restrict__ q, long total) {   // over DH=8
  long i = (long)blockIdx.x * blockDim.x + threadIdx.x;
  if (i >= total) return;
  float* p = q + (i >> 3) * EMB_ + (i & 7) * 8;
  float mx = p[0];
#pragma unroll
  for (int d = 1; d < 8; ++d) mx = fmaxf(mx, p[d]);
  float e[8], s = 0.f;
#pragma unroll
  for (int d = 0; d < 8; ++d) { e[d] = expf(p[d] - mx); s += e[d]; }
  float inv = 1.f / s;
#pragma unroll
  for (int d = 0; d < 8; ++d) p[d] = e[d] * inv;
}

__global__ void softmax_seq(float* __restrict__ k, int T) {         // over T per (b,c)
  __shared__ float red[256];
  const int bc = blockIdx.x, bb = bc >> 6, c = bc & 63;
  float* base = k + (long)bb * T * EMB_ + c;
  float mx = -3.4e38f;
  for (int t = threadIdx.x; t < T; t += 256) mx = fmaxf(mx, base[(long)t * EMB_]);
  red[threadIdx.x] = mx; __syncthreads();
  for (int st = 128; st > 0; st >>= 1) {
    if (threadIdx.x < st) red[threadIdx.x] = fmaxf(red[threadIdx.x], red[threadIdx.x + st]);
    __syncthreads();
  }
  mx = red[0]; __syncthreads();
  float s = 0.f;
  for (int t = threadIdx.x; t < T; t += 256) s += expf(base[(long)t * EMB_] - mx);
  red[threadIdx.x] = s; __syncthreads();
  for (int st = 128; st > 0; st >>= 1) {
    if (threadIdx.x < st) red[threadIdx.x] += red[threadIdx.x + st];
    __syncthreads();
  }
  const float inv = 1.f / red[0];
  for (int t = threadIdx.x; t < T; t += 256) {
    long o = (long)t * EMB_;
    base[o] = expf(base[o] - mx) * inv;
  }
}

__global__ __launch_bounds__(64) void lin_ctx(
    const float* __restrict__ k, const float* __restrict__ v,
    float* __restrict__ ctx, int T)
{
  const int bh = blockIdx.x, bb = bh >> 3, h = bh & 7;
  const int d = threadIdx.x >> 3, e = threadIdx.x & 7;
  const float* kp = k + (long)bb * T * EMB_ + h * 8 + d;
  const float* vp = v + (long)bb * T * EMB_ + h * 8 + e;
  float acc = 0.f;
  for (int t = 0; t < T; ++t) acc += kp[(long)t * EMB_] * vp[(long)t * EMB_];
  ctx[(long)bh * 64 + d * 8 + e] = acc;
}

__global__ void lin_apply(const float* __restrict__ q, const float* __restrict__ ctx,
                          float* __restrict__ o, int T, long total)
{
  long i = (long)blockIdx.x * blockDim.x + threadIdx.x;
  if (i >= total) return;
  const long n = i >> 6; const int c = (int)(i & 63), h = c >> 3, e = c & 7;
  const int bb = (int)(n / T);
  const float* qp = q + n * EMB_ + h * 8;
  const float* cp = ctx + ((long)bb * 8 + h) * 64 + e;
  float acc = 0.f;
#pragma unroll
  for (int d = 0; d < 8; ++d) acc += qp[d] * cp[d * 8];
  o[i] = acc;
}

__global__ void add_pe(float* __restrict__ dst, const float* __restrict__ src,
                       int T, long total)
{
  long i = (long)blockIdx.x * blockDim.x + threadIdx.x;
  if (i >= total) return;
  const long n = i >> 6; const int c = (int)(i & 63);
  const int t = (int)(n % T);
  const float freq = expf(-(float)(c & ~1) * (LOG10K_ / EMB_));
  const float ang = (float)t * freq;
  dst[i] = src[i] + ((c & 1) ? cosf(ang) : sinf(ang));
}

// ---------------------------- weight reshaping ------------------------------
__global__ void pack_c1(const float* __restrict__ w, float* __restrict__ p) {
  int i = blockIdx.x * 256 + threadIdx.x; if (i >= 256 * 64) return;
  int k = i >> 6, co = i & 63;
  p[i] = w[co * 256 + k];                       // (64,1,256) -> [256][64]
}
__global__ void pack_c2(const float* __restrict__ w, float* __restrict__ p) {
  int i = blockIdx.x * 256 + threadIdx.x; if (i >= 256 * 64) return;
  int co = i & 63, pp = i >> 6, ci = pp & 63, j = pp >> 6;
  p[i] = w[co * 256 + ci * 4 + j];              // patch pos (j,ci) -> [256][64]
}
__global__ void pack_d2(const float* __restrict__ w, float* __restrict__ p) {
  int i = blockIdx.x * 256 + threadIdx.x; if (i >= 4 * 64 * 64) return;
  int j = i >> 12, rem = i & 4095, ci = rem >> 6, co = rem & 63;
  p[i] = w[ci * 256 + co * 4 + j];              // (Cin,Cout,4) -> per-tap [64][64]
}
__global__ void cb_halfnorm(const float* __restrict__ cbk, float* __restrict__ cn) {
  int c = blockIdx.x * 256 + threadIdx.x; if (c >= CB_) return;
  const float* p = cbk + (long)c * EMB_;
  float s = 0.f;
#pragma unroll
  for (int k = 0; k < EMB_; ++k) s += p[k] * p[k];
  cn[c] = 0.5f * s;
}
__global__ void gather_codes(const float* __restrict__ cbk, const int* __restrict__ idx,
                             float* __restrict__ out, long total)
{
  long i = (long)blockIdx.x * blockDim.x + threadIdx.x;
  if (i >= total) return;
  out[i] = cbk[(long)idx[i >> 6] * EMB_ + (i & 63)];
}

// ConvTranspose1d EMB->1, k=256, s=16, then GELU. 16 taps x 64 chan per output.
__global__ void deconv1_gelu(const float* __restrict__ hin, const float* __restrict__ w,
                             const float* __restrict__ bias, float* __restrict__ out,
                             long total)
{
  long i = (long)blockIdx.x * blockDim.x + threadIdx.x;
  if (i >= total) return;
  const int bb = (int)(i / TOUT_), to = (int)(i % TOUT_);
  const int jb = to & 15, tbase = to >> 4;
  float acc = bias[0];
  for (int m = 0; m < 16; ++m) {
    int ti = tbase - m;
    if (ti < 0 || ti >= T3_) continue;
    const float* hp = hin + ((long)bb * T3_ + ti) * EMB_;
    const float* wp = w + (jb + 16 * m);        // d1_w[ci][0][j] = w[ci*256 + j]
    float a = 0.f;
#pragma unroll
    for (int ci = 0; ci < EMB_; ++ci) a += hp[ci] * wp[ci * 256];
    acc += a;
  }
  out[i] = gelu_f(acc);
}

// ----------------------------- host-side glue -------------------------------
static inline void launch_gemm(hipStream_t s, const float* A, const _Float16* Bp,
                               const float* bias, const float* resid, float* C,
                               int Mrows, int N, int K, int ldc,
                               int a_mul, int a_extra, int rpb,
                               int out_mul, int out_add, int act_gelu)
{
  dim3 grid((unsigned)(Mrows / 16), (unsigned)(N / 16));
  gemm16<<<grid, 32, 0, s>>>(A, Bp, bias, resid, C, K, ldc,
                             a_mul, a_extra, rpb, out_mul, out_add, act_gelu);
}

// packed per-layer layout: Wq(4096) Wk(4096) Wv(4096) Wo(4096) W1(16384) W2(16384)
static void tf_layer(hipStream_t s, float* h, float* tY, float* tQ, float* tK,
                     float* tV, float* tO, float* tFF, float* ctx,
                     const float* const* W, const _Float16* packed, int l)
{
  const float* ln1w = W[0] + l * 64;   const float* ln1b = W[1] + l * 64;
  const float* ln2w = W[6] + l * 64;   const float* ln2b = W[7] + l * 64;
  const float* b1 = W[9] + l * 256;    const float* b2 = W[11] + l * 64;
  const _Float16* pw = packed + (size_t)l * LSTRIDE_;

  layernorm_row<<<NTOK_, 32, 0, s>>>(h, tY, ln1w, ln1b);
  launch_gemm(s, tY, pw,         nullptr, nullptr, tQ, NTOK_, 64, 64, 64, 64, 0, 1, 1, 0, 0);
  launch_gemm(s, tY, pw + 4096,  nullptr, nullptr, tK, NTOK_, 64, 64, 64, 64, 0, 1, 1, 0, 0);
  launch_gemm(s, tY, pw + 8192,  nullptr, nullptr, tV, NTOK_, 64, 64, 64, 64, 0, 1, 1, 0, 0);
  softmax_head<<<(NTOK_ * 8 + 255) / 256, 256, 0, s>>>(tQ, (long)NTOK_ * 8);
  softmax_seq<<<B_ * 64, 256, 0, s>>>(tK, T2_);
  lin_ctx<<<B_ * 8, 64, 0, s>>>(tK, tV, ctx, T2_);
  lin_apply<<<(NTOK_ * 64 + 255) / 256, 256, 0, s>>>(tQ, ctx, tO, T2_, (long)NTOK_ * 64);
  launch_gemm(s, tO, pw + 12288, nullptr, h, h, NTOK_, 64, 64, 64, 64, 0, 1, 1, 0, 0);
  layernorm_row<<<NTOK_, 32, 0, s>>>(h, tY, ln2w, ln2b);
  launch_gemm(s, tY, pw + 16384, b1, nullptr, tFF, NTOK_, 256, 64, 256, 64, 0, 1, 1, 0, 1);
  launch_gemm(s, tFF, pw + 32768, b2, h, h, NTOK_, 64, 256, 64, 256, 0, 1, 1, 0, 0);
}

extern "C" void kernel_launch(void* const* d_in, const int* in_sizes, int n_in,
                              void* d_out, int out_size, void* d_ws, size_t ws_size,
                              hipStream_t stream)
{
  (void)in_sizes; (void)n_in; (void)out_size; (void)ws_size;

  const float* x    = (const float*)d_in[0];
  const float* c1w  = (const float*)d_in[1];  const float* c1b  = (const float*)d_in[2];
  const float* gn1w = (const float*)d_in[3];  const float* gn1b = (const float*)d_in[4];
  const float* c2w  = (const float*)d_in[5];  const float* c2b  = (const float*)d_in[6];
  const float* gn2w = (const float*)d_in[7];  const float* gn2b = (const float*)d_in[8];
  const float* TE[12]; for (int i = 0; i < 12; ++i) TE[i] = (const float*)d_in[9 + i];
  const float* cbk  = (const float*)d_in[21];
  const float* TD[12]; for (int i = 0; i < 12; ++i) TD[i] = (const float*)d_in[22 + i];
  const float* d2w  = (const float*)d_in[34]; const float* d2bias = (const float*)d_in[35];
  const float* gnw  = (const float*)d_in[36]; const float* gnb    = (const float*)d_in[37];
  const float* d1w  = (const float*)d_in[38]; const float* d1bias = (const float*)d_in[39];

  // outputs: h [16,1,65520] | idx [16,1020] | qo [16,1020,64] | quant_in [same]
  float* out_h   = (float*)d_out;
  float* out_idx = out_h + (long)B_ * TOUT_;
  float* out_qo  = out_idx + NTOK_;
  float* out_qin = out_qo + (long)NTOK_ * EMB_;

  // workspace bump allocator (32-byte aligned chunks for b128/v16h loads)
  float* ws = (float*)d_ws; size_t off = 0;
  auto alloc = [&](size_t n) { float* p = ws + off; off += (n + 7) & ~(size_t)7; return p; };
  auto alloch = [&](size_t nh) { return (_Float16*)alloc((nh + 1) / 2); };

  float* h1   = alloc((size_t)N1_ * EMB_);        // conv1 out / reused for deconv2 out
  float* hA   = alloc((size_t)NTOK_ * EMB_);
  float* tY   = alloc((size_t)NTOK_ * EMB_);
  float* tQ   = alloc((size_t)NTOK_ * EMB_);
  float* tK   = alloc((size_t)NTOK_ * EMB_);
  float* tV   = alloc((size_t)NTOK_ * EMB_);
  float* tO   = alloc((size_t)NTOK_ * EMB_);
  float* tFF  = alloc((size_t)NTOK_ * FF_);
  float* decH = alloc((size_t)NTOK_ * EMB_);
  float* ctx  = alloc((size_t)B_ * 8 * 64);
  float* cbn  = alloc(CB_);
  float* pk1  = alloc(256 * 64);
  float* pk2  = alloc(256 * 64);
  float* pkD2 = alloc(4 * 64 * 64);
  int*   idxi = (int*)alloc(NTOK_);
  _Float16* packE = alloch((size_t)12 * LSTRIDE_);  // encoder stack, f16 fragments
  _Float16* packD = alloch((size_t)3 * LSTRIDE_);   // decoder stack
  _Float16* pk1h  = alloch(256 * 64);
  _Float16* pk2h  = alloch(256 * 64);
  _Float16* pkD2h = alloch(4 * 64 * 64);
  _Float16* cbh   = alloch((size_t)EMB_ * CB_);     // packed codebook

  // --- weight reshape + f16 fragment packing + codebook half-norms ---
  pack_c1<<<64, 256, 0, stream>>>(c1w, pk1);
  pack_c2<<<64, 256, 0, stream>>>(c2w, pk2);
  pack_d2<<<64, 256, 0, stream>>>(d2w, pkD2);
  cb_halfnorm<<<16, 256, 0, stream>>>(cbk, cbn);
  pack_b16<<<64, 256, 0, stream>>>(pk1, pk1h, 256, 64, 0);
  pack_b16<<<64, 256, 0, stream>>>(pk2, pk2h, 256, 64, 0);
  for (int j = 0; j < 4; ++j)
    pack_b16<<<16, 256, 0, stream>>>(pkD2 + j * 4096, pkD2h + j * 4096, 64, 64, 0);
  pack_b16<<<1024, 256, 0, stream>>>(cbk, cbh, 64, 4096, 1);  // B[k][c] = cb[c][k]
  for (int l = 0; l < 12; ++l) {
    _Float16* base = packE + (size_t)l * LSTRIDE_;
    pack_b16<<<16, 256, 0, stream>>>(TE[2] + l * 4096, base,          64, 64, 0);
    pack_b16<<<16, 256, 0, stream>>>(TE[3] + l * 4096, base + 4096,   64, 64, 0);
    pack_b16<<<16, 256, 0, stream>>>(TE[4] + l * 4096, base + 8192,   64, 64, 0);
    pack_b16<<<16, 256, 0, stream>>>(TE[5] + l * 4096, base + 12288,  64, 64, 0);
    pack_b16<<<64, 256, 0, stream>>>(TE[8] + l * 16384, base + 16384, 64, 256, 0);
    pack_b16<<<64, 256, 0, stream>>>(TE[10] + l * 16384, base + 32768, 256, 64, 0);
  }
  for (int l = 0; l < 3; ++l) {
    _Float16* base = packD + (size_t)l * LSTRIDE_;
    pack_b16<<<16, 256, 0, stream>>>(TD[2] + l * 4096, base,          64, 64, 0);
    pack_b16<<<16, 256, 0, stream>>>(TD[3] + l * 4096, base + 4096,   64, 64, 0);
    pack_b16<<<16, 256, 0, stream>>>(TD[4] + l * 4096, base + 8192,   64, 64, 0);
    pack_b16<<<16, 256, 0, stream>>>(TD[5] + l * 4096, base + 12288,  64, 64, 0);
    pack_b16<<<64, 256, 0, stream>>>(TD[8] + l * 16384, base + 16384, 64, 256, 0);
    pack_b16<<<64, 256, 0, stream>>>(TD[10] + l * 16384, base + 32768, 256, 64, 0);
  }

  // --- encoder convs (as implicit-im2col WMMA GEMMs) ---
  // conv1: row r=(b,t) -> x + 16r + 240b ; K=256
  launch_gemm(stream, x, pk1h, c1b, nullptr, h1, N1_, 64, 256, 64,
              16, 240, T1_, 1, 0, 1);
  groupnorm_tok<<<B_ * 16, 256, 0, stream>>>(h1, gn1w, gn1b, T1_);
  // conv2: 4 contiguous token rows form the 256-wide patch; +64 float skew per batch
  launch_gemm(stream, h1, pk2h, c2b, nullptr, hA, NTOK_, 64, 256, 64,
              256, 64, T2_, 1, 0, 1);
  groupnorm_tok<<<B_ * 16, 256, 0, stream>>>(hA, gn2w, gn2b, T2_);

  const long tot64 = (long)NTOK_ * EMB_;
  add_pe<<<(unsigned)((tot64 + 255) / 256), 256, 0, stream>>>(hA, hA, T2_, tot64);

  for (int l = 0; l < 12; ++l)
    tf_layer(stream, hA, tY, tQ, tK, tV, tO, tFF, ctx, TE, packE, l);

  // --- vector quantization ---
  l2norm_row<<<NTOK_, 32, 0, stream>>>(hA, out_qin);
  vq_argmin<<<NTOK_ / 16, 32, 0, stream>>>(out_qin, cbh, cbn, idxi, out_idx);
  gather_codes<<<(unsigned)((tot64 + 255) / 256), 256, 0, stream>>>(cbk, idxi, out_qo, tot64);

  // --- decoder transformer (input = qo + pe; straight-through fwd == codebook[idx]) ---
  add_pe<<<(unsigned)((tot64 + 255) / 256), 256, 0, stream>>>(decH, out_qo, T2_, tot64);
  for (int l = 0; l < 3; ++l)
    tf_layer(stream, decH, tY, tQ, tK, tV, tO, tFF, ctx, TD, packD, l);

  // --- deconv2 (stride==kernel==4): one GEMM per tap, interleaved output rows ---
  float* d2buf = h1;  // 16*4080*64 <= 16*4081*64, h1 is dead now
  for (int j = 0; j < 4; ++j)
    launch_gemm(stream, decH, pkD2h + j * 4096, d2bias, nullptr, d2buf,
                NTOK_, 64, 64, 64, 64, 0, 1, 4, j, 1);
  groupnorm_tok<<<B_ * 16, 256, 0, stream>>>(d2buf, gnw, gnb, T3_);

  // --- deconv1 + gelu -> final output ---
  const long totO = (long)B_ * TOUT_;
  deconv1_gelu<<<(unsigned)((totO + 255) / 256), 256, 0, stream>>>(d2buf, d1w, d1bias, out_h, totO);
}
